// GRUAttentionDecoder_21277267984619
// MI455X (gfx1250) — matrix operand order repaired and checked by reading, not compiled, not run on hardware
//
#include <hip/hip_runtime.h>
#include <hip/hip_bf16.h>

typedef __attribute__((ext_vector_type(16))) __bf16 v16bf;
typedef __attribute__((ext_vector_type(8)))  __bf16 v8bf;
typedef __attribute__((ext_vector_type(4)))  __bf16 v4bf;
typedef __attribute__((ext_vector_type(8)))  float  v8f;

// ---------------------------------------------------------------------------
// Generic WMMA GEMM: Y[M,N] = act( A1[M,K1] * B1[N,K1]^T (+ A2*B2^T) + bias + addend )
// A row-major (M,K) bf16, B row-major (N,K) bf16 (i.e. we compute X * W^T).
// One wave computes one 16x16 tile; 4 waves per block cover 64 N-columns.
// Fragment layouts per CDNA5 ISA 7.12.2 (wave32):
//   A 16x32 bf16 : lane L -> row m=L&15, halves [kb8..kb8+7] and [kb8+16..kb8+23], kb8 = (L&16)?8:0
//   B 32x16 bf16 : lane L -> col n=L&15, halves [kb8..kb8+7] and [kb8+16..kb8+23], same k order as A
//   C/D 16x16 f32: lane L -> col n=L&15, rows mBase..mBase+7, mBase = (L&16)?8:0
// ---------------------------------------------------------------------------
__device__ inline void wmma_acc_k(v8f& c, const __bf16* __restrict__ A,
                                  const __bf16* __restrict__ Bw, int K,
                                  int mA, int nB, int kA8, int kB16) {
  const __bf16* arow = A + (long)mA * K;
  const __bf16* brow = Bw + (long)nB * K;
  for (int k0 = 0; k0 < K; k0 += 32) {
    v8bf alo = *(const v8bf*)(arow + k0 + kA8);
    v8bf ahi = *(const v8bf*)(arow + k0 + kA8 + 16);
    v16bf a = __builtin_shufflevector(alo, ahi, 0, 1, 2, 3, 4, 5, 6, 7,
                                      8, 9, 10, 11, 12, 13, 14, 15);
    v8bf blo = *(const v8bf*)(brow + k0 + kA8);
    v8bf bhi = *(const v8bf*)(brow + k0 + kA8 + 16);
    v16bf b = __builtin_shufflevector(blo, bhi, 0, 1, 2, 3, 4, 5, 6, 7,
                                      8, 9, 10, 11, 12, 13, 14, 15);
    c = __builtin_amdgcn_wmma_f32_16x16x32_bf16(false, a, false, b,
                                                (short)0, c, false, false);
  }
}

__global__ __launch_bounds__(128) void wmma_gemm_kernel(
    const __bf16* __restrict__ A1, const __bf16* __restrict__ B1, int K1,
    const __bf16* __restrict__ A2, const __bf16* __restrict__ B2, int K2,
    const float* __restrict__ bias, const float* __restrict__ addend,
    float* __restrict__ Yf, __bf16* __restrict__ Ybf,
    int N, int act) {
  const int lane  = threadIdx.x & 31;
  const int wave  = threadIdx.x >> 5;
  const int tileN = blockIdx.x * 64 + wave * 16;
  const int tileM = blockIdx.y * 16;

  const int mA   = tileM + (lane & 15);
  const int nB   = tileN + (lane & 15);
  const int kA8  = (lane & 16) ? 8 : 0;
  const int kB16 = (lane & 16) ? 16 : 0;

  v8f c = {0.f, 0.f, 0.f, 0.f, 0.f, 0.f, 0.f, 0.f};
  wmma_acc_k(c, A1, B1, K1, mA, nB, kA8, kB16);
  if (A2) wmma_acc_k(c, A2, B2, K2, mA, nB, kA8, kB16);

  const int mBase = tileM + ((lane & 16) ? 8 : 0);
  const float bv = bias ? bias[nB] : 0.0f;
#pragma unroll
  for (int r = 0; r < 8; ++r) {
    long idx = (long)(mBase + r) * N + nB;
    float v = c[r] + bv;
    if (addend) v += addend[idx];
    if (act == 1)      v = 1.0f / (1.0f + __expf(-v));   // sigmoid
    else if (act == 2) v = tanhf(v);                     // tanh
    if (Yf)  Yf[idx]  = v;
    if (Ybf) Ybf[idx] = (__bf16)v;
  }
}

// ---------------------------------------------------------------------------
// f32 -> bf16 conversion (weights / activations staged for WMMA operands)
// ---------------------------------------------------------------------------
__global__ void cvt_bf16_kernel(const float* __restrict__ s, __bf16* __restrict__ d, long n) {
  long i = (long)blockIdx.x * 256 + threadIdx.x;
  if (i < n) d[i] = (__bf16)s[i];
}

__global__ void inith_kernel(const float* __restrict__ s, float* __restrict__ hf,
                             __bf16* __restrict__ hb, int n) {
  int i = blockIdx.x * 256 + threadIdx.x;
  if (i < n) { float v = s[i]; hf[i] = v; hb[i] = (__bf16)v; }
}

// rh = h_prev * sigmoid_gates[:, :H]  (bf16 for the next WMMA)
__global__ void rh_kernel(const float* __restrict__ gates, const float* __restrict__ hprev,
                          __bf16* __restrict__ rh_bf, int H) {
  int i = blockIdx.x * 256 + threadIdx.x;  // i < B*H
  int b = i / H, j = i - b * H;
  rh_bf[i] = (__bf16)(hprev[i] * gates[b * 2 * H + j]);
}

// h = ym*(u*hprev + (1-u)*cand) + (1-ym)*hprev ; u = gates[:, H:2H]
__global__ void gate_kernel(const float* __restrict__ gates, const float* __restrict__ hprev,
                            const float* __restrict__ cand, const float* __restrict__ ym,
                            float* __restrict__ hout, __bf16* __restrict__ hout_bf,
                            float* __restrict__ hs_out, float* __restrict__ ss_out, int H) {
  int i = blockIdx.x * 256 + threadIdx.x;  // i < B*H
  int b = i / H, j = i - b * H;
  float u  = gates[b * 2 * H + H + j];
  float hp = hprev[i];
  float h  = u * hp + (1.0f - u) * cand[i];
  float m  = ym[b];
  h = m * h + (1.0f - m) * hp;
  hout[i]    = h;
  hout_bf[i] = (__bf16)h;
  if (hs_out) { hs_out[i] = h; ss_out[i] = h; }
}

// e[tx,b] = xmask[tx,b] * sum_c tanh(pctx_bf[tx,b,c] + hatt[b,c]) * U_att[c]
// one wave per (tx*B + b); 8 waves / block. Streams 52MB of bf16 pctx from L2.
__global__ __launch_bounds__(256) void energy_kernel(
    const __bf16* __restrict__ pctx, const float* __restrict__ hatt,
    const float* __restrict__ uatt, const float* __restrict__ xmask,
    float* __restrict__ e, int B, int C) {
  const int lane = threadIdx.x & 31;
  const int wave = threadIdx.x >> 5;
  const long txb = (long)blockIdx.x * 8 + wave;
  const int b = (int)(txb % B);
  const __bf16* row = pctx + txb * C;
  const float* hrow = hatt + (long)b * C;
  float acc = 0.0f;
  for (int c0 = lane * 4; c0 < C; c0 += 128) {
    __builtin_prefetch(row + c0 + 512, 0, 1);  // global_prefetch_b8
    v4bf   p4 = *(const v4bf*)(row + c0);
    float4 h4 = *(const float4*)(hrow + c0);
    float4 u4 = *(const float4*)(uatt + c0);
    acc += tanhf((float)p4[0] + h4.x) * u4.x;
    acc += tanhf((float)p4[1] + h4.y) * u4.y;
    acc += tanhf((float)p4[2] + h4.z) * u4.z;
    acc += tanhf((float)p4[3] + h4.w) * u4.w;
  }
#pragma unroll
  for (int m = 16; m >= 1; m >>= 1) acc += __shfl_xor(acc, m, 32);
  if (lane == 0) e[txb] = acc * xmask[txb];
}

// softmax over Tx per batch column b; also writes dists[t][b][tx]
__global__ __launch_bounds__(256) void softmax_kernel(
    const float* __restrict__ e, const float* __restrict__ xmask,
    float* __restrict__ a, float* __restrict__ dists_t, int Tx, int B) {
  const int b = blockIdx.x;
  __shared__ float red[256];
  float mx = -1e30f;
  for (int tx = threadIdx.x; tx < Tx; tx += 256) mx = fmaxf(mx, e[(long)tx * B + b]);
  red[threadIdx.x] = mx;
  __syncthreads();
  for (int s = 128; s > 0; s >>= 1) {
    if (threadIdx.x < s) red[threadIdx.x] = fmaxf(red[threadIdx.x], red[threadIdx.x + s]);
    __syncthreads();
  }
  mx = red[0];
  __syncthreads();
  float sum = 0.0f;
  for (int tx = threadIdx.x; tx < Tx; tx += 256) {
    float v = __expf(e[(long)tx * B + b] - mx) * xmask[(long)tx * B + b];
    a[(long)tx * B + b] = v;
    sum += v;
  }
  red[threadIdx.x] = sum;
  __syncthreads();
  for (int s = 128; s > 0; s >>= 1) {
    if (threadIdx.x < s) red[threadIdx.x] += red[threadIdx.x + s];
    __syncthreads();
  }
  float inv = 1.0f / red[0];
  for (int tx = threadIdx.x; tx < Tx; tx += 256) {
    float v = a[(long)tx * B + b] * inv;
    a[(long)tx * B + b] = v;
    dists_t[(long)b * Tx + tx] = v;
  }
}

// atted[b,c] = sum_tx a[tx,b]*ctx[tx,b,c]; also writes atts[t][b][c] and bf16 copy
__global__ __launch_bounds__(256) void attctx_kernel(
    const float* __restrict__ a, const float* __restrict__ ctx,
    float* __restrict__ atted, __bf16* __restrict__ atted_bf,
    float* __restrict__ atts_t, int Tx, int B, int C) {
  const int b = blockIdx.y;
  const int c = blockIdx.x * 256 + threadIdx.x;
  float acc = 0.0f;
  for (int tx = 0; tx < Tx; ++tx)
    acc += a[(long)tx * B + b] * ctx[((long)tx * B + b) * C + c];
  long idx = (long)b * C + c;
  atted[idx]    = acc;
  atted_bf[idx] = (__bf16)acc;
  atts_t[idx]   = acc;
}

// xids[t][b][tx] = (float)xid[tx][b]
__global__ void xids_kernel(const int* __restrict__ xid, float* __restrict__ out,
                            int Ty, int B, int Tx) {
  long i = (long)blockIdx.x * 256 + threadIdx.x;
  long tot = (long)Ty * B * Tx;
  if (i < tot) {
    int tx = (int)(i % Tx);
    long tb = i / Tx;
    int b = (int)(tb % B);
    out[i] = (float)xid[(long)tx * B + b];
  }
}

// ---------------------------------------------------------------------------
extern "C" void kernel_launch(void* const* d_in, const int* in_sizes, int n_in,
                              void* d_out, int out_size, void* d_ws, size_t ws_size,
                              hipStream_t stream) {
  constexpr int Ty = 100, Tx = 400, B = 64, I = 512, H = 512, C = 1024, H2 = 2 * H;

  const float* y_emb      = (const float*)d_in[0];
  const float* context    = (const float*)d_in[1];
  const float* init_state = (const float*)d_in[2];
  const float* x_mask     = (const float*)d_in[3];
  const float* y_mask     = (const float*)d_in[4];
  const int*   xid        = (const int*)d_in[5];
  const float* W          = (const float*)d_in[6];
  const float* U          = (const float*)d_in[7];
  const float* b          = (const float*)d_in[8];
  const float* Wx         = (const float*)d_in[9];
  const float* Ux         = (const float*)d_in[10];
  const float* bx         = (const float*)d_in[11];
  const float* Wc_att     = (const float*)d_in[12];
  const float* b_att      = (const float*)d_in[13];
  const float* W_comb_att = (const float*)d_in[14];
  const float* U_att      = (const float*)d_in[15];
  const float* U_nl       = (const float*)d_in[16];
  const float* b_nl       = (const float*)d_in[17];
  const float* Ux_nl      = (const float*)d_in[18];
  const float* bx_nl      = (const float*)d_in[19];
  const float* Wc         = (const float*)d_in[20];
  const float* Wcx        = (const float*)d_in[21];

  float* out   = (float*)d_out;
  float* hs    = out;
  float* ss    = hs + (long)Ty * B * H;
  float* atts  = ss + (long)Ty * B * H;
  float* dists = atts + (long)Ty * B * C;
  float* xids  = dists + (long)Ty * B * Tx;

  // ---- carve workspace (256B aligned chunks) ----
  char* wp = (char*)d_ws;
  auto carve = [&](size_t bytes) -> void* {
    void* r = (void*)wp;
    wp += (bytes + 255) & ~(size_t)255;
    return r;
  };
  __bf16* ctx_bf   = (__bf16*)carve((size_t)Tx * B * C * 2);
  __bf16* yemb_bf  = (__bf16*)carve((size_t)Ty * B * I * 2);
  __bf16* pctx_bf  = (__bf16*)carve((size_t)Tx * B * C * 2);
  float*  x        = (float*)carve((size_t)Ty * B * H2 * 4);
  float*  xx       = (float*)carve((size_t)Ty * B * H * 4);
  __bf16* Wcatt_bf = (__bf16*)carve((size_t)C * C * 2);
  __bf16* W_bf     = (__bf16*)carve((size_t)H2 * I * 2);
  __bf16* Wx_bf    = (__bf16*)carve((size_t)H * I * 2);
  __bf16* U_bf     = (__bf16*)carve((size_t)H2 * H * 2);
  __bf16* Ux_bf    = (__bf16*)carve((size_t)H * H * 2);
  __bf16* Wcomb_bf = (__bf16*)carve((size_t)C * H * 2);
  __bf16* Unl_bf   = (__bf16*)carve((size_t)H2 * H * 2);
  __bf16* Uxnl_bf  = (__bf16*)carve((size_t)H * H * 2);
  __bf16* Wc_bf    = (__bf16*)carve((size_t)H2 * C * 2);
  __bf16* Wcx_bf   = (__bf16*)carve((size_t)H * C * 2);
  float*  hprev    = (float*)carve((size_t)B * H * 4);
  __bf16* hprev_bf = (__bf16*)carve((size_t)B * H * 2);
  float*  gates1   = (float*)carve((size_t)B * H2 * 4);
  __bf16* rh_bf    = (__bf16*)carve((size_t)B * H * 2);
  float*  cand     = (float*)carve((size_t)B * H * 4);
  float*  h1       = (float*)carve((size_t)B * H * 4);
  __bf16* h1_bf    = (__bf16*)carve((size_t)B * H * 2);
  float*  hatt     = (float*)carve((size_t)B * C * 4);
  float*  evec     = (float*)carve((size_t)Tx * B * 4);
  float*  avec     = (float*)carve((size_t)Tx * B * 4);
  float*  atted    = (float*)carve((size_t)B * C * 4);
  __bf16* atted_bf = (__bf16*)carve((size_t)B * C * 2);
  float*  gates2   = (float*)carve((size_t)B * H2 * 4);
  __bf16* rh2_bf   = (__bf16*)carve((size_t)B * H * 2);

  auto cvt = [&](const float* s, __bf16* d, long n) {
    cvt_bf16_kernel<<<(n + 255) / 256, 256, 0, stream>>>(s, d, n);
  };
  // bf16 staging of activations + weights
  cvt(context, ctx_bf, (long)Tx * B * C);
  cvt(y_emb, yemb_bf, (long)Ty * B * I);
  cvt(Wc_att, Wcatt_bf, (long)C * C);
  cvt(W, W_bf, (long)H2 * I);
  cvt(Wx, Wx_bf, (long)H * I);
  cvt(U, U_bf, (long)H2 * H);
  cvt(Ux, Ux_bf, (long)H * H);
  cvt(W_comb_att, Wcomb_bf, (long)C * H);
  cvt(U_nl, Unl_bf, (long)H2 * H);
  cvt(Ux_nl, Uxnl_bf, (long)H * H);
  cvt(Wc, Wc_bf, (long)H2 * C);
  cvt(Wcx, Wcx_bf, (long)H * C);

  // pctx (bf16 only: 52MB, L2-resident across all 100 steps)
  wmma_gemm_kernel<<<dim3(C / 64, (Tx * B) / 16), 128, 0, stream>>>(
      ctx_bf, Wcatt_bf, C, nullptr, nullptr, 0, b_att, nullptr,
      nullptr, pctx_bf, C, 0);
  // x = y_emb @ W^T + b ; xx = y_emb @ Wx^T + bx
  wmma_gemm_kernel<<<dim3(H2 / 64, (Ty * B) / 16), 128, 0, stream>>>(
      yemb_bf, W_bf, I, nullptr, nullptr, 0, b, nullptr, x, nullptr, H2, 0);
  wmma_gemm_kernel<<<dim3(H / 64, (Ty * B) / 16), 128, 0, stream>>>(
      yemb_bf, Wx_bf, I, nullptr, nullptr, 0, bx, nullptr, xx, nullptr, H, 0);

  inith_kernel<<<(B * H + 255) / 256, 256, 0, stream>>>(init_state, hprev, hprev_bf, B * H);

  const int ebh = (B * H + 255) / 256;
  for (int t = 0; t < Ty; ++t) {
    const float* x_t  = x + (long)t * B * H2;
    const float* xx_t = xx + (long)t * B * H;
    const float* ym_t = y_mask + (long)t * B;

    // GRU-1: gates1 = sigmoid(hprev U^T + x_t)
    wmma_gemm_kernel<<<dim3(H2 / 64, B / 16), 128, 0, stream>>>(
        hprev_bf, U_bf, H, nullptr, nullptr, 0, nullptr, x_t, gates1, nullptr, H2, 1);
    rh_kernel<<<ebh, 256, 0, stream>>>(gates1, hprev, rh_bf, H);
    // cand = tanh((r*hprev) Ux^T + xx_t)
    wmma_gemm_kernel<<<dim3(H / 64, B / 16), 128, 0, stream>>>(
        rh_bf, Ux_bf, H, nullptr, nullptr, 0, nullptr, xx_t, cand, nullptr, H, 2);
    gate_kernel<<<ebh, 256, 0, stream>>>(gates1, hprev, cand, ym_t, h1, h1_bf,
                                         nullptr, nullptr, H);

    // attention: hatt = h1 W_comb^T ; e ; softmax ; weighted context
    wmma_gemm_kernel<<<dim3(C / 64, B / 16), 128, 0, stream>>>(
        h1_bf, Wcomb_bf, H, nullptr, nullptr, 0, nullptr, nullptr, hatt, nullptr, C, 0);
    energy_kernel<<<(Tx * B) / 8, 256, 0, stream>>>(pctx_bf, hatt, U_att, x_mask,
                                                    evec, B, C);
    softmax_kernel<<<B, 256, 0, stream>>>(evec, x_mask, avec,
                                          dists + (long)t * B * Tx, Tx, B);
    attctx_kernel<<<dim3(C / 256, B), 256, 0, stream>>>(
        avec, context, atted, atted_bf, atts + (long)t * B * C, Tx, B, C);

    // GRU-2 (fused dual GEMM): gates2 = sigmoid(atted Wc^T + h1 U_nl^T + b_nl)
    wmma_gemm_kernel<<<dim3(H2 / 64, B / 16), 128, 0, stream>>>(
        atted_bf, Wc_bf, C, h1_bf, Unl_bf, H, b_nl, nullptr, gates2, nullptr, H2, 1);
    rh_kernel<<<ebh, 256, 0, stream>>>(gates2, h1, rh2_bf, H);
    // cand = tanh(atted Wcx^T + (r2*h1) Ux_nl^T + bx_nl)
    wmma_gemm_kernel<<<dim3(H / 64, B / 16), 128, 0, stream>>>(
        atted_bf, Wcx_bf, C, rh2_bf, Uxnl_bf, H, bx_nl, nullptr, cand, nullptr, H, 2);
    gate_kernel<<<ebh, 256, 0, stream>>>(gates2, h1, cand, ym_t, hprev, hprev_bf,
                                         hs + (long)t * B * H, ss + (long)t * B * H, H);
  }

  long ntot = (long)Ty * B * Tx;
  xids_kernel<<<(ntot + 255) / 256, 256, 0, stream>>>(xid, xids, Ty, B, Tx);
}
